// MyRNN_42468636623203
// MI455X (gfx1250) — compile-verified, matrix operand order checked by reading, not run on hardware
//
#include <hip/hip_runtime.h>
#include <hip/hip_bf16.h>

typedef __bf16 bf16_t;
typedef __attribute__((ext_vector_type(16))) __bf16 v16bf;
typedef __attribute__((ext_vector_type(8)))  __bf16 v8bf;
typedef __attribute__((ext_vector_type(8)))  float  v8f;

#define BATCH  4096
#define TSTEPS 80
#define EDIM   100
#define VOCAB  10000
#define UNITS  512
#define GATES  2048   // 4*UNITS
#define KC     64     // k-chunk per LDS stage (2 WMMA k-steps)

__device__ __forceinline__ float sigmoidf_(float x) {
    return 1.0f / (1.0f + __expf(-x));
}

// ---------------------------------------------------------------------------
// CDNA5 async global->LDS copy (16B per lane), tracked by ASYNCcnt.
// VDST VGPR carries the LDS byte address, VADDR the 64-bit global address.
// ---------------------------------------------------------------------------
__device__ __forceinline__ void async_copy_b128(unsigned lds_byte_off, const void* gptr) {
    asm volatile("global_load_async_to_lds_b128 %0, %1, off"
                 :: "v"(lds_byte_off), "v"(gptr) : "memory");
}
__device__ __forceinline__ void wait_async0() {
    asm volatile("s_wait_asynccnt 0x0" ::: "memory");
}

// ---------------------------------------------------------------------------
// fp32 [512][2048] -> bf16 pre-transposed [2048][512] ( [g*512+n][k], k-contig )
// ---------------------------------------------------------------------------
__global__ void cvt_transpose_kernel(const float* __restrict__ in, bf16_t* __restrict__ out) {
    size_t idx = (size_t)blockIdx.x * 256 + threadIdx.x;   // over GATES*UNITS
    int c = (int)(idx >> 9);       // column of original (gate*512+n)
    int k = (int)(idx & 511);      // k
    out[idx] = (bf16_t)in[(size_t)k * GATES + c];
}

// ---------------------------------------------------------------------------
// Zero a float region (also covers bf16 buffers via byte count / 4)
// ---------------------------------------------------------------------------
__global__ void zero_kernel(float* __restrict__ p, size_t n) {
    size_t i = (size_t)blockIdx.x * 256 + threadIdx.x;
    if (i < n) p[i] = 0.0f;
}

// ---------------------------------------------------------------------------
// embW[v, c] = sum_k emb[v,k] * W0[k,c]   (one-time, K=100, trivially cheap)
// ---------------------------------------------------------------------------
__global__ void embw_kernel(const float* __restrict__ emb, const float* __restrict__ W0,
                            bf16_t* __restrict__ embW) {
    size_t idx = (size_t)blockIdx.x * 256 + threadIdx.x;   // over VOCAB*GATES
    int v = (int)(idx >> 11);
    int c = (int)(idx & 2047);
    const float* e = emb + (size_t)v * EDIM;
    float s = 0.0f;
    #pragma unroll 4
    for (int k = 0; k < EDIM; ++k) s += e[k] * W0[(size_t)k * GATES + c];
    embW[idx] = (bf16_t)s;
}

// ---------------------------------------------------------------------------
// Build a 16x32 bf16 fragment from an LDS row: two contiguous 8-elem K-runs
// (ISA 7.12.2: lanes 0-15 hold K 0-7 & 16-23; lanes 16-31 hold K 8-15 & 24-31)
// ---------------------------------------------------------------------------
__device__ __forceinline__ v16bf load_frag(const bf16_t* row, int half) {
    v8bf lo = *reinterpret_cast<const v8bf*>(row + 8 * half);
    v8bf hi = *reinterpret_cast<const v8bf*>(row + 16 + 8 * half);
    return __builtin_shufflevector(lo, hi, 0, 1, 2, 3, 4, 5, 6, 7,
                                           8, 9, 10, 11, 12, 13, 14, 15);
}

// ---------------------------------------------------------------------------
// Fused LSTM step:  z = A1@B1 (+ A2@B2) (+ embW gather) + bias, then gates.
// Block = 256 threads (8 waves): 32 rows x 64 units x 4 gates.
// Grid = (BATCH/32, UNITS/64). B matrices pre-transposed: [g*512+n][k].
// ---------------------------------------------------------------------------
__global__ __launch_bounds__(256) void lstm_step_kernel(
    const bf16_t* __restrict__ A1, const bf16_t* __restrict__ B1t,
    const bf16_t* __restrict__ A2, const bf16_t* __restrict__ B2t,
    const float*  __restrict__ bias,
    const bf16_t* __restrict__ embW, const int* __restrict__ tokens, int t,
    float* __restrict__ c_state, bf16_t* __restrict__ h_out)
{
    // k-contiguous tiles; row strides of 144B (36 dwords) -> conflict-free b128
    __shared__ bf16_t Atile[32][KC + 8];        //  4.6 KB
    __shared__ bf16_t Btile[4][64][KC + 8];     // 36.9 KB

    const int tid  = threadIdx.x;
    const int lane = tid & 31;
    const int wid  = tid >> 5;
    const int half = lane >> 4;
    const int l16  = lane & 15;
    const int wm   = (wid >> 2) << 4;           // 0 or 16
    const int wn   = (wid & 3)  << 4;           // 0,16,32,48

    const int m0 = blockIdx.x * 32;
    const int u0 = blockIdx.y * 64;

    v8f acc[4] = {};                            // i, f, g, o accumulators

    const int npass = (A2 != nullptr) ? 2 : 1;
    for (int pass = 0; pass < npass; ++pass) {
        const bf16_t* A  = pass ? A2 : A1;
        const bf16_t* Bt = pass ? B2t : B1t;

        // per-thread tile-copy coordinates
        const int ar = tid >> 3, as = (tid & 7) * 8;      // A: row, 16B segment
        const int bg = tid >> 6, bn = tid & 63;           // B: gate, n-row
        const bf16_t* aSrcRow = A  + (size_t)(m0 + ar) * UNITS;
        const bf16_t* bSrcRow = Bt + (size_t)(bg * UNITS + u0 + bn) * UNITS;
        const unsigned aDst = (unsigned)(size_t)&Atile[ar][as];
        const unsigned bDst = (unsigned)(size_t)&Btile[bg][bn][0];

        for (int kc = 0; kc < UNITS / KC; ++kc) {
            const int k0 = kc * KC;
            // A tile: 32 rows x 64 k, one async 16B copy per thread
            async_copy_b128(aDst, aSrcRow + k0 + as);
            // B tile: 4 gates x 64 n-rows x 64 k, 8 async 16B copies (contig)
            #pragma unroll
            for (int s = 0; s < 8; ++s)
                async_copy_b128(bDst + s * 16, bSrcRow + k0 + s * 8);
            // prefetch next chunk toward L2 while async engine fills LDS
            if (kc + 1 < UNITS / KC) {
                __builtin_prefetch(aSrcRow + k0 + KC + as, 0, 1);
                __builtin_prefetch(bSrcRow + k0 + KC, 0, 1);
            }
            wait_async0();
            __syncthreads();

            #pragma unroll
            for (int ks = 0; ks < KC / 32; ++ks) {
                v16bf afrag = load_frag(&Atile[wm + l16][ks * 32], half);
                #pragma unroll
                for (int g = 0; g < 4; ++g) {
                    v16bf bfrag = load_frag(&Btile[g][wn + l16][ks * 32], half);
                    acc[g] = __builtin_amdgcn_wmma_f32_16x16x32_bf16(
                        false, afrag, false, bfrag, (short)0, acc[g], false, false);
                }
            }
            __syncthreads();
        }
    }

    // Epilogue: gate math + state update. D layout: m = r + 8*half, n = l16.
    #pragma unroll
    for (int r = 0; r < 8; ++r) {
        int m = m0 + wm + r + 8 * half;
        int u = u0 + wn + l16;
        float zi = acc[0][r] + bias[u];
        float zf = acc[1][r] + bias[UNITS + u];
        float zg = acc[2][r] + bias[2 * UNITS + u];
        float zo = acc[3][r] + bias[3 * UNITS + u];
        if (embW != nullptr) {
            int tok = tokens[(size_t)m * TSTEPS + t];
            const bf16_t* e = embW + (size_t)tok * GATES;
            zi += (float)e[u];
            zf += (float)e[UNITS + u];
            zg += (float)e[2 * UNITS + u];
            zo += (float)e[3 * UNITS + u];
        }
        size_t off = (size_t)m * UNITS + u;
        float c_old = c_state[off];
        float c_new = sigmoidf_(zf) * c_old + sigmoidf_(zi) * tanhf(zg);
        c_state[off] = c_new;
        h_out[off]   = (bf16_t)(sigmoidf_(zo) * tanhf(c_new));
    }
}

// ---------------------------------------------------------------------------
// out[m] = sigmoid( h1[m,:] . Wout + bout ) ; one wave per row
// ---------------------------------------------------------------------------
__global__ __launch_bounds__(256) void out_kernel(const bf16_t* __restrict__ h1,
                                                  const float* __restrict__ Wout,
                                                  const float* __restrict__ bout,
                                                  float* __restrict__ out) {
    int row  = blockIdx.x * 8 + (threadIdx.x >> 5);
    int lane = threadIdx.x & 31;
    float s = 0.0f;
    #pragma unroll
    for (int k = lane; k < UNITS; k += 32)
        s += (float)h1[(size_t)row * UNITS + k] * Wout[k];
    #pragma unroll
    for (int off = 16; off; off >>= 1) s += __shfl_down(s, off, 32);
    if (lane == 0) out[row] = sigmoidf_(s + bout[0]);
}

// ---------------------------------------------------------------------------
extern "C" void kernel_launch(void* const* d_in, const int* in_sizes, int n_in,
                              void* d_out, int out_size, void* d_ws, size_t ws_size,
                              hipStream_t stream)
{
    (void)in_sizes; (void)n_in; (void)out_size; (void)ws_size;

    const int*   inputs = (const int*)  d_in[0];
    const float* emb    = (const float*)d_in[1];
    const float* W0     = (const float*)d_in[2];
    const float* U0     = (const float*)d_in[3];
    const float* b0     = (const float*)d_in[4];
    const float* W1     = (const float*)d_in[5];
    const float* U1     = (const float*)d_in[6];
    const float* b1     = (const float*)d_in[7];
    const float* Wout   = (const float*)d_in[8];
    const float* bout   = (const float*)d_in[9];
    float* out = (float*)d_out;

    // Workspace layout (~79 MB total)
    char* p = (char*)d_ws;
    bf16_t* U0b  = (bf16_t*)p; p += (size_t)UNITS * GATES * 2;
    bf16_t* W1b  = (bf16_t*)p; p += (size_t)UNITS * GATES * 2;
    bf16_t* U1b  = (bf16_t*)p; p += (size_t)UNITS * GATES * 2;
    bf16_t* embW = (bf16_t*)p; p += (size_t)VOCAB * GATES * 2;
    bf16_t* h0[2], *h1[2];
    h0[0] = (bf16_t*)p; p += (size_t)BATCH * UNITS * 2;
    h0[1] = (bf16_t*)p; p += (size_t)BATCH * UNITS * 2;
    h1[0] = (bf16_t*)p; p += (size_t)BATCH * UNITS * 2;
    h1[1] = (bf16_t*)p; p += (size_t)BATCH * UNITS * 2;
    float* c0 = (float*)p; p += (size_t)BATCH * UNITS * 4;
    float* c1 = (float*)p; p += (size_t)BATCH * UNITS * 4;

    // 1) Convert recurrent/input weights to bf16, transposed to [g*512+n][k]
    const int wtot = UNITS * GATES;
    cvt_transpose_kernel<<<wtot / 256, 256, 0, stream>>>(U0, U0b);
    cvt_transpose_kernel<<<wtot / 256, 256, 0, stream>>>(W1, W1b);
    cvt_transpose_kernel<<<wtot / 256, 256, 0, stream>>>(U1, U1b);

    // 2) Precompute embW = emb @ W0 (folds embedding + input projection)
    embw_kernel<<<(VOCAB * GATES) / 256, 256, 0, stream>>>(emb, W0, embW);

    // 3) Zero h0/h1 (both buffers) and c0/c1 — contiguous region after embW
    size_t zfloats = ((size_t)BATCH * UNITS * 2 * 4
                    + (size_t)BATCH * UNITS * 4 * 2) / 4;
    zero_kernel<<<(unsigned)((zfloats + 255) / 256), 256, 0, stream>>>((float*)h0[0], zfloats);

    // 4) 80 sequential steps, 2 fused layer kernels each
    dim3 grid(BATCH / 32, UNITS / 64), blk(256);
    for (int t = 0; t < TSTEPS; ++t) {
        int cur = t & 1, nxt = (t + 1) & 1;
        lstm_step_kernel<<<grid, blk, 0, stream>>>(
            h0[cur], U0b, nullptr, nullptr, b0, embW, inputs, t, c0, h0[nxt]);
        lstm_step_kernel<<<grid, blk, 0, stream>>>(
            h0[nxt], W1b, h1[cur], U1b, b1, nullptr, nullptr, t, c1, h1[nxt]);
    }

    // 5) Output head (final h1 lives in buffer (TSTEPS)%2 == 0)
    out_kernel<<<BATCH / 8, 256, 0, stream>>>(h1[0], Wout, bout, out);
}
